// AdmEdgeDetect_38534446579778
// MI455X (gfx1250) — matrix operand order repaired and checked by reading, not compiled, not run on hardware
//
#include <hip/hip_runtime.h>

typedef __attribute__((ext_vector_type(16))) _Float16 v16h;
typedef __attribute__((ext_vector_type(8)))  _Float16 v8h;
typedef __attribute__((ext_vector_type(4)))  _Float16 v4h;
typedef __attribute__((ext_vector_type(8)))  float    v8f;

#define HIMG 1024
#define WIMG 1024
#define TILE 32
#define INROWS 48
#define INCOLS 48
#define MIDROWS 56   // 48 mid rows stored at +4, pad rows 52..55 zeroed

#if __has_builtin(__builtin_amdgcn_global_load_async_to_lds_b32)
#define HAVE_ASYNC_LDS 1
#else
#define HAVE_ASYNC_LDS 0
#endif

typedef __attribute__((address_space(1))) int g_i32;
typedef __attribute__((address_space(3))) int l_i32;

__device__ __forceinline__ void wait_asynccnt0() {
#if __has_builtin(__builtin_amdgcn_s_wait_asynccnt)
  __builtin_amdgcn_s_wait_asynccnt(0);
#else
  asm volatile("s_wait_asynccnt 0" ::: "memory");
#endif
}

// ---------------------------------------------------------------------------
// Prep kernel: extract separable 1-D factors from the 2-D reference filters
// and pre-bake the banded weight matrices directly in WMMA fragment layout.
//   filters[2s] = fx_s = u_s(x) (x) v_s(y):
//     u_s[k] = sum_y fx[y][k]   (column sums),  v_s[k] = sum_x |fx[k][x]|
// bandH[f][lane][16]: B-operand (32x16) of horizontal conv with taps f1d[f]
// bandV[f][lane][16]: A-operand (16x32) of vertical conv with taps f1d[f^1]
// ---------------------------------------------------------------------------
__global__ void prep_filters(const float* __restrict__ filt,
                             _Float16* __restrict__ bandH,
                             _Float16* __restrict__ bandV) {
  __shared__ float sf[72];
  int t = threadIdx.x;
  if (t < 72) {
    int s = t / 18, r = t % 18;
    const float* fx = filt + (2 * s) * 81;
    float acc = 0.f;
    if (r < 9) { int k = r;     for (int y = 0; y < 9; ++y) acc += fx[y * 9 + k]; }
    else       { int k = r - 9; for (int x = 0; x < 9; ++x) acc += fabsf(fx[k * 9 + x]); }
    sf[t] = acc;
  }
  __syncthreads();
  int f = t >> 5, lane = t & 31;
  int n = lane & 15, hi = lane >> 4;
  const float* hH = &sf[f * 9];
  const float* hV = &sf[(f ^ 1) * 9];
  for (int j = 0; j < 16; ++j) {
    int kb = (j < 8) ? (8 * hi + j) : (16 + 8 * hi + (j - 8));  // B: table order
    int tH = kb - n;
    bandH[(f * 32 + lane) * 16 + j] =
        (tH >= 0 && tH <= 8) ? (_Float16)hH[tH] : (_Float16)0.f;
    int k = 16 * hi + j;                                    // A: K = 16*hi + j
    int tV = k - n;
    bandV[(f * 32 + lane) * 16 + j] =
        (tV >= 0 && tV <= 8) ? (_Float16)hV[tV] : (_Float16)0.f;
  }
}

// ---------------------------------------------------------------------------
// Main fused kernel: one 32x32 output tile per 256-thread (8-wave) workgroup.
// Wave wv owns filter wv. 6 horizontal + 4 vertical WMMAs per wave, paired
// so one job's f32->f16 converts fill the other's WMMA hazard window.
// ---------------------------------------------------------------------------
__global__ __launch_bounds__(256) void edge_kernel(
    const float* __restrict__ x,
    const _Float16* __restrict__ bandHg,
    const _Float16* __restrict__ bandVg,
    const float* __restrict__ basep, const float* __restrict__ up,
    const float* __restrict__ lp, float* __restrict__ outG,
    float* __restrict__ outW)
{
#if HAVE_ASYNC_LDS
  __shared__ alignas(16) float    sInF[INROWS * INCOLS];    // async DMA target
#endif
  __shared__ alignas(16) _Float16 sIn[INROWS * INCOLS];     // input halo (f16)
  __shared__ alignas(16) _Float16 sMidT[8][32 * MIDROWS];   // mid, transposed
  __shared__ alignas(16) _Float16 sGT[8][32 * 32];          // gx/gy, transposed

  const int tid  = threadIdx.x;
  const int img  = blockIdx.z;
  const int R0   = blockIdx.y * TILE;
  const int C0   = blockIdx.x * TILE;
  const float* in = x + (size_t)img * HIMG * WIMG;

  const int wv   = tid >> 5;
  const int lane = tid & 31;
  const int nrow = lane & 15;
  const int hi   = lane >> 4;

  // ---- Stage input tile rows R0-8..R0+39, cols C0-4..C0+43 (circular wrap).
  { const int ty = tid >> 4, tx = tid & 15;
#pragma unroll
    for (int a = 0; a < 3; ++a) {
      int tr = ty + 16 * a;
      int gr = (R0 - 8 + tr) & (HIMG - 1);
      const float* row = in + gr * WIMG;
#pragma unroll
      for (int b = 0; b < 3; ++b) {
        int tc = tx + 16 * b;
        int gc = (C0 - 4 + tc) & (WIMG - 1);
#if HAVE_ASYNC_LDS
        // ASYNCcnt-tracked DMA: no VGPR return, wave keeps issuing.
        __builtin_amdgcn_global_load_async_to_lds_b32(
            (g_i32*)(row + gc),
            (l_i32*)&sInF[tr * INCOLS + tc],
            0, 0);
#else
        sIn[tr * INCOLS + tc] = (_Float16)row[gc];
#endif
      }
    }
  }

  // Per-wave banded weight fragments (32B/lane) overlap the tile fill.
  const v16h bH = *(const v16h*)&bandHg[(wv * 32 + lane) * 16];
  const v16h bV = *(const v16h*)&bandVg[(wv * 32 + lane) * 16];

  // Zero the 4 pad rows (52..55) of the transposed mid buffer.
  *(v4h*)&sMidT[wv][lane * MIDROWS + 52] = (v4h)0;

#if HAVE_ASYNC_LDS
  wait_asynccnt0();
  __syncthreads();
  // LDS->LDS f32 -> f16 conversion pass.
#pragma unroll
  for (int e = 0; e < 9; ++e) {
    int i = tid + 256 * e;
    sIn[i] = (_Float16)sInF[i];
  }
#endif
  __syncthreads();

  // ---- Horizontal pass: 3 paired WMMAs (row-block x {cb=0, cb=16}).
#pragma unroll
  for (int rblk = 0; rblk < 3; ++rblk) {
    const int rb = rblk * 16;           // mid row block (row 0 = R0-8)
    const _Float16* pr = &sIn[(rb + nrow) * INCOLS];
    v8h a0 = *(const v8h*)&pr[8 * hi];          // K 0..7 / 8..15   (cb=0)
    v8h a1 = *(const v8h*)&pr[16 + 8 * hi];     // shared middle run
    v8h a2 = *(const v8h*)&pr[32 + 8 * hi];     // K 16..23 / 24..31 (cb=16)
    v16h A0 = __builtin_shufflevector(a0, a1, 0,1,2,3,4,5,6,7,
                                              8,9,10,11,12,13,14,15);
    v16h A1 = __builtin_shufflevector(a1, a2, 0,1,2,3,4,5,6,7,
                                              8,9,10,11,12,13,14,15);
    v8f z = {};
    v8f d0 = __builtin_amdgcn_wmma_f32_16x16x32_f16(false, A0, false, bH,
                                                    (short)0, z, false, false);
    v8f d1 = __builtin_amdgcn_wmma_f32_16x16x32_f16(false, A1, false, bH,
                                                    (short)0, z, false, false);
    v8h h0, h1;
#pragma unroll
    for (int i = 0; i < 8; ++i) { h0[i] = (_Float16)d0[i]; h1[i] = (_Float16)d1[i]; }
    // D rows M=i+8*hi -> transposed store, rows shifted by +4.
    _Float16* p0 = &sMidT[wv][nrow * MIDROWS + rb + 4 + 8 * hi];
    _Float16* p1 = &sMidT[wv][(16 + nrow) * MIDROWS + rb + 4 + 8 * hi];
    *(v4h*)p0       = __builtin_shufflevector(h0, h0, 0, 1, 2, 3);
    *(v4h*)(p0 + 4) = __builtin_shufflevector(h0, h0, 4, 5, 6, 7);
    *(v4h*)p1       = __builtin_shufflevector(h1, h1, 0, 1, 2, 3);
    *(v4h*)(p1 + 4) = __builtin_shufflevector(h1, h1, 4, 5, 6, 7);
  }
  __syncthreads();

  // ---- Vertical pass: 2 paired WMMAs x 2 (band in A, mid in B).
#pragma unroll
  for (int cblk = 0; cblk < 2; ++cblk) {
    const int cb = cblk * 16;
    const _Float16* pc = &sMidT[wv][(cb + nrow) * MIDROWS + 8 + 16 * hi];
    v8h b0 = *(const v8h*)pc;                 // r0=0:  rows  8+16hi..
    v8h b1 = *(const v8h*)(pc + 8);
    v8h b2 = *(const v8h*)(pc + 16);          // r0=16: rows 24+16hi..
    v8h b3 = *(const v8h*)(pc + 24);
    v16h B0 = __builtin_shufflevector(b0, b1, 0,1,2,3,4,5,6,7,
                                              8,9,10,11,12,13,14,15);
    v16h B1 = __builtin_shufflevector(b2, b3, 0,1,2,3,4,5,6,7,
                                              8,9,10,11,12,13,14,15);
    v8f z = {};
    v8f d0 = __builtin_amdgcn_wmma_f32_16x16x32_f16(false, bV, false, B0,
                                                    (short)0, z, false, false);
    v8f d1 = __builtin_amdgcn_wmma_f32_16x16x32_f16(false, bV, false, B1,
                                                    (short)0, z, false, false);
    v8h h0, h1;
#pragma unroll
    for (int i = 0; i < 8; ++i) { h0[i] = (_Float16)d0[i]; h1[i] = (_Float16)d1[i]; }
    *(v8h*)&sGT[wv][(cb + nrow) * 32 + 8 * hi]      = h0;   // 16B aligned
    *(v8h*)&sGT[wv][(cb + nrow) * 32 + 16 + 8 * hi] = h1;
  }
  __syncthreads();

  // ---- Fused magnitude / max over scales / power threshold; coalesced stores.
  const float base = basep[0], ut = up[0], lt = lp[0];
  const float lb = __logf(base);
  const size_t HW = (size_t)HIMG * WIMG;
  const int cc = tid & 31, rbase = tid >> 5;
#pragma unroll
  for (int i = 0; i < 4; ++i) {
    int r = rbase + 8 * i;
    float g = 0.f;
#pragma unroll
    for (int s = 0; s < 4; ++s) {
      float gx = (float)sGT[2 * s][cc * 32 + r];
      float gy = (float)sGT[2 * s + 1][cc * 32 + r];
      g = fmaxf(g, __fsqrt_rn(gx * gx + gy * gy));
    }
    float w = __expf(g * lb) - 1.f;   // base^g - 1
    w = (w > ut) ? 1.f : w;
    w = (w < lt) ? 0.f : w;
    size_t o = (size_t)img * HW + (size_t)(R0 + r) * WIMG + (C0 + cc);
    outG[o] = g;
    outW[o] = w;
  }
}

extern "C" void kernel_launch(void* const* d_in, const int* in_sizes, int n_in,
                              void* d_out, int out_size, void* d_ws, size_t ws_size,
                              hipStream_t stream) {
  const float* x       = (const float*)d_in[0];
  const float* filters = (const float*)d_in[1];
  const float* base    = (const float*)d_in[2];
  const float* ut      = (const float*)d_in[3];
  const float* lt      = (const float*)d_in[4];
  // d_in[5]=idx, d_in[6]=ite: multiplied by 0.0 in the reference -> unused.

  int Bimg = in_sizes[0] / (HIMG * WIMG);
  _Float16* bandH = (_Float16*)((char*)d_ws + 512);           // 8*32*16 halfs
  _Float16* bandV = bandH + 8 * 32 * 16;
  prep_filters<<<1, 256, 0, stream>>>(filters, bandH, bandV);

  float* outG = (float*)d_out;
  float* outW = outG + (size_t)Bimg * HIMG * WIMG;
  dim3 grid(WIMG / TILE, HIMG / TILE, Bimg);
  edge_kernel<<<grid, 256, 0, stream>>>(x, bandH, bandV, base, ut, lt,
                                        outG, outW);
}
